// APPNP__36215164240761
// MI455X (gfx1250) — compile-verified, matrix operand order checked by reading, not run on hardware
//
#include <hip/hip_runtime.h>
#include <hip/hip_bf16.h>

typedef __attribute__((ext_vector_type(16))) _Float16 v16h;
typedef __attribute__((ext_vector_type(8)))  _Float16 v8h;
typedef __attribute__((ext_vector_type(4)))  _Float16 v4h;
typedef __attribute__((ext_vector_type(8)))  float    v8f;

#define F_IN  512
#define HID   256
#define CLS   40
#define CPAD  48
#define ALPHA 0.1f
#define KSTEPS 10

#define SA 40   // LDS stride (halves) for A tile rows -> 80B, 16B aligned
#define SH 264  // LDS stride (halves) for h1 bounce rows -> 528B, 16B aligned

__device__ __forceinline__ v16h cat16(v8h a, v8h b) {
    return __builtin_shufflevector(a, b, 0,1,2,3,4,5,6,7,8,9,10,11,12,13,14,15);
}
__device__ __forceinline__ v8f wmma16(v16h a, v16h b, v8f c) {
    return __builtin_amdgcn_wmma_f32_16x16x32_f16(false, a, false, b, (short)0, c, false, false);
}
__device__ __forceinline__ v4h cvt4(float4 v) {
    v4h r; r[0] = (_Float16)v.x; r[1] = (_Float16)v.y;
    r[2] = (_Float16)v.z; r[3] = (_Float16)v.w; return r;
}

// ---- one-time weight conversion ------------------------------------------
__global__ void convW1_kernel(const float* __restrict__ W1, _Float16* __restrict__ W1T) {
    int i = blockIdx.x * blockDim.x + threadIdx.x;       // over 512*256, coalesced on n
    if (i < F_IN * HID) {
        int k = i >> 8, n = i & 255;
        W1T[(size_t)n * F_IN + k] = (_Float16)W1[i];     // col-major f16 [n][k]
    }
}
__global__ void convW2_kernel(const float* __restrict__ W2, _Float16* __restrict__ W2T) {
    int i = blockIdx.x * blockDim.x + threadIdx.x;       // over 48*256
    if (i < CPAD * HID) {
        int c = i % CPAD, k = i / CPAD;
        W2T[(size_t)c * HID + k] = (c < CLS) ? (_Float16)W2[(size_t)k * CLS + c]
                                             : (_Float16)0.f;
    }
}

// ---- GEMM1: h1 = relu(x @ W1 + b1), f16 out, [N][256] --------------------
// 128 threads = 4 waves; WG tile 32(M) x 256(N); wave w owns 64 cols, 2 M-subtiles.
__global__ __launch_bounds__(128) void gemm1_kernel(
    const float* __restrict__ x, const _Float16* __restrict__ W1T,
    const float* __restrict__ b1, _Float16* __restrict__ h1, int N)
{
    __shared__ alignas(16) _Float16 lA[32 * SA];     // x tile 32x32 f16
    __shared__ alignas(16) _Float16 lOut[32 * SH];   // h1 bounce 32x256 f16

    const int tid  = threadIdx.x;
    const int w    = tid >> 5;
    const int lane = tid & 31;
    const int lo = lane & 15, hi = lane >> 4;
    const size_t m0 = (size_t)blockIdx.x * 32;

    // A staging: 32x32 elems = 256 float4; thread handles 2 (pipelined)
    const int f0 = tid, f1 = tid + 128;
    const int r0 = f0 >> 3, c0 = (f0 & 7) * 4;
    const int r1 = f1 >> 3, c1 = (f1 & 7) * 4;
    const size_t row0 = (m0 + r0 < (size_t)N) ? m0 + r0 : (size_t)N - 1;
    const size_t row1 = (m0 + r1 < (size_t)N) ? m0 + r1 : (size_t)N - 1;

    float4 ra0 = *(const float4*)&x[row0 * F_IN + c0];
    float4 ra1 = *(const float4*)&x[row1 * F_IN + c1];

    v8f acc[2][4] = {};

    for (int kc = 0; kc < F_IN; kc += 32) {
        *(v4h*)&lA[r0 * SA + c0] = cvt4(ra0);
        *(v4h*)&lA[r1 * SA + c1] = cvt4(ra1);
        __syncthreads();
        if (kc + 32 < F_IN) {               // prefetch next x chunk during WMMAs
            ra0 = *(const float4*)&x[row0 * F_IN + kc + 32 + c0];
            ra1 = *(const float4*)&x[row1 * F_IN + kc + 32 + c1];
        }
        // hoist ALL fragment loads: 8 global b128 (B) + 4 ds b128 (A) per chunk
        v16h bf[4];
        #pragma unroll
        for (int t = 0; t < 4; ++t) {
            const _Float16* bp = W1T + (size_t)(w * 64 + t * 16 + lo) * F_IN + kc;
            bf[t] = cat16(*(const v8h*)(bp + hi * 8), *(const v8h*)(bp + 16 + hi * 8));
        }
        v16h af[2];
        #pragma unroll
        for (int s = 0; s < 2; ++s) {
            v8h a0 = *(const v8h*)&lA[(s * 16 + lo) * SA + hi * 8];
            v8h a1 = *(const v8h*)&lA[(s * 16 + lo) * SA + 16 + hi * 8];
            af[s] = cat16(a0, a1);
        }
        #pragma unroll
        for (int t = 0; t < 4; ++t) {
            acc[0][t] = wmma16(af[0], bf[t], acc[0][t]);
            acc[1][t] = wmma16(af[1], bf[t], acc[1][t]);
        }
        __syncthreads();
    }

    // epilogue: bias + relu -> f16 bounce -> coalesced b128 store to h1
    #pragma unroll
    for (int t = 0; t < 4; ++t) {
        int n = w * 64 + t * 16 + lo;       // C layout: lane->N, vgpr r -> M=r+hi*8
        float bias = b1[n];
        #pragma unroll
        for (int s = 0; s < 2; ++s)
            #pragma unroll
            for (int r = 0; r < 8; ++r) {
                float v = acc[s][t][r] + bias;
                lOut[(s * 16 + r + hi * 8) * SH + n] = (_Float16)(v > 0.f ? v : 0.f);
            }
    }
    __syncthreads();
    for (int i = tid; i < 32 * 32; i += 128) {           // 32 rows x 32 chunks(16B)
        int r = i >> 5, c8 = (i & 31) * 8;
        if (m0 + r < (size_t)N)
            *(v8h*)&h1[(m0 + r) * HID + c8] = *(const v8h*)&lOut[r * SH + c8];
    }
}

// ---- GEMM2: h = h1 @ W2 + b2, f32 out, [N][40] ---------------------------
// 128 threads = 4 waves; wave w owns rows m0+16w; A and B fragments direct global.
__global__ __launch_bounds__(128) void gemm2_kernel(
    const _Float16* __restrict__ h1, const _Float16* __restrict__ W2T,
    const float* __restrict__ b2, float* __restrict__ h, int N)
{
    __shared__ alignas(16) float lOut[64 * CPAD];

    const int tid  = threadIdx.x;
    const int w    = tid >> 5;
    const int lane = tid & 31;
    const int lo = lane & 15, hi = lane >> 4;
    const size_t m0 = (size_t)blockIdx.x * 64;
    const size_t mw = m0 + (size_t)w * 16;

    v8f acc[3] = {};
    for (int kc = 0; kc < HID; kc += 32) {
        const _Float16* ap = h1 + (mw + lo) * HID + kc;   // rows beyond N: slack
        v16h bf[3];
        #pragma unroll
        for (int t = 0; t < 3; ++t) {
            const _Float16* bp = W2T + (size_t)(t * 16 + lo) * HID + kc;
            bf[t] = cat16(*(const v8h*)(bp + hi * 8), *(const v8h*)(bp + 16 + hi * 8));
        }
        v16h af = cat16(*(const v8h*)(ap + hi * 8), *(const v8h*)(ap + 16 + hi * 8));
        #pragma unroll
        for (int t = 0; t < 3; ++t)
            acc[t] = wmma16(af, bf[t], acc[t]);
    }
    #pragma unroll
    for (int t = 0; t < 3; ++t) {
        int c = t * 16 + lo;
        float bias = (c < CLS) ? b2[c] : 0.f;
        #pragma unroll
        for (int r = 0; r < 8; ++r)
            lOut[(w * 16 + r + hi * 8) * CPAD + c] = acc[t][r] + bias;
    }
    __syncthreads();
    for (int i = tid; i < 64 * 10; i += 128) {            // 64 rows x 10 float4
        int r = i / 10, q = i % 10;
        size_t m = m0 + r;
        if (m < (size_t)N)
            *(float4*)&h[m * CLS + q * 4] = *(const float4*)&lOut[r * CPAD + q * 4];
    }
}

// ---------------- graph preprocessing --------------------------------------
__global__ void zero_i32(int* __restrict__ p, int n) {
    int i = blockIdx.x * blockDim.x + threadIdx.x;
    if (i < n) p[i] = 0;
}

__global__ void count_kernel(const int* __restrict__ row, int* __restrict__ cnt, int E) {
    int e = blockIdx.x * blockDim.x + threadIdx.x;
    if (e < E) atomicAdd(&cnt[row[e]], 1);
}

__global__ void dinv_kernel(const int* __restrict__ cnt, float* __restrict__ dinv, int n) {
    int i = blockIdx.x * blockDim.x + threadIdx.x;
    if (i < n) dinv[i] = rsqrtf((float)cnt[i] + 1.0f);   // +1 self loop
}

__global__ __launch_bounds__(1024) void scan_kernel(
    const int* __restrict__ cnt, int* __restrict__ rowptr, int n)
{
    __shared__ int buf[1024];
    __shared__ int carry;
    const int tid = threadIdx.x;
    if (tid == 0) carry = 0;
    __syncthreads();
    for (int base = 0; base < n; base += 1024) {
        int i = base + tid;
        int v = (i < n) ? cnt[i] : 0;
        buf[tid] = v;
        __syncthreads();
        #pragma unroll
        for (int off = 1; off < 1024; off <<= 1) {
            int t = (tid >= off) ? buf[tid - off] : 0;
            __syncthreads();
            buf[tid] += t;
            __syncthreads();
        }
        if (i < n) rowptr[i] = carry + buf[tid] - v;
        __syncthreads();
        if (tid == 1023) carry += buf[1023];
        __syncthreads();
    }
    if (tid == 0) rowptr[n] = carry;
}

__global__ void fill_kernel(const int* __restrict__ row, const int* __restrict__ col,
                            const int* __restrict__ rowptr, int* __restrict__ fill,
                            const float* __restrict__ dinv,
                            int* __restrict__ ecol, float* __restrict__ ew, int E)
{
    int e = blockIdx.x * blockDim.x + threadIdx.x;
    if (e < E) {
        int r = row[e], c = col[e];
        int pos = rowptr[r] + atomicAdd(&fill[r], 1);
        ecol[pos] = c;
        ew[pos]   = dinv[r] * dinv[c];
    }
}

__global__ void copy_f32(const float* __restrict__ src, float* __restrict__ dst, int n) {
    int i = blockIdx.x * blockDim.x + threadIdx.x;
    if (i < n) dst[i] = src[i];
}

// ---- one APPNP step (CSR gather, 40 VGPR accumulators) --------------------
__global__ __launch_bounds__(256) void step_kernel(
    const int* __restrict__ rowptr, const int* __restrict__ ecol,
    const float* __restrict__ ew, const float* __restrict__ dinv,
    const float* __restrict__ h, const float* __restrict__ zin,
    float* __restrict__ zout, int n)
{
    int i = blockIdx.x * blockDim.x + threadIdx.x;
    if (i >= n) return;

    const int js = rowptr[i];
    const int jend = rowptr[i + 1];
    __builtin_prefetch(&ew[js], 0, 3);        // global_prefetch for edge stream
    __builtin_prefetch(&ecol[js], 0, 3);

    float acc[CLS];
    const float4* zi = (const float4*)(zin + (size_t)i * CLS);
    float dw = dinv[i] * dinv[i];
    #pragma unroll
    for (int q = 0; q < CLS / 4; ++q) {
        float4 v = zi[q];
        acc[q * 4 + 0] = dw * v.x; acc[q * 4 + 1] = dw * v.y;
        acc[q * 4 + 2] = dw * v.z; acc[q * 4 + 3] = dw * v.w;
    }
    for (int j = js; j < jend; ++j) {
        float wgt = ew[j];
        const float4* zc = (const float4*)(zin + (size_t)ecol[j] * CLS);
        #pragma unroll
        for (int q = 0; q < CLS / 4; ++q) {
            float4 v = zc[q];
            acc[q * 4 + 0] += wgt * v.x; acc[q * 4 + 1] += wgt * v.y;
            acc[q * 4 + 2] += wgt * v.z; acc[q * 4 + 3] += wgt * v.w;
        }
    }
    const float4* hp = (const float4*)(h + (size_t)i * CLS);
    float4* zo = (float4*)(zout + (size_t)i * CLS);
    #pragma unroll
    for (int q = 0; q < CLS / 4; ++q) {
        float4 hv = hp[q];
        float4 o;
        o.x = (1.f - ALPHA) * acc[q * 4 + 0] + ALPHA * hv.x;
        o.y = (1.f - ALPHA) * acc[q * 4 + 1] + ALPHA * hv.y;
        o.z = (1.f - ALPHA) * acc[q * 4 + 2] + ALPHA * hv.z;
        o.w = (1.f - ALPHA) * acc[q * 4 + 3] + ALPHA * hv.w;
        zo[q] = o;
    }
}

__global__ void logsoftmax_kernel(const float* __restrict__ z, float* __restrict__ out, int n) {
    int i = blockIdx.x * blockDim.x + threadIdx.x;
    if (i >= n) return;
    float v[CLS];
    const float4* zi = (const float4*)(z + (size_t)i * CLS);
    #pragma unroll
    for (int q = 0; q < CLS / 4; ++q) {
        float4 t = zi[q];
        v[q * 4 + 0] = t.x; v[q * 4 + 1] = t.y; v[q * 4 + 2] = t.z; v[q * 4 + 3] = t.w;
    }
    float m = v[0];
    #pragma unroll
    for (int c = 1; c < CLS; ++c) m = fmaxf(m, v[c]);
    float s = 0.f;
    #pragma unroll
    for (int c = 0; c < CLS; ++c) s += __expf(v[c] - m);
    float ls = __logf(s) + m;
    float4* oi = (float4*)(out + (size_t)i * CLS);
    #pragma unroll
    for (int q = 0; q < CLS / 4; ++q) {
        float4 o;
        o.x = v[q * 4 + 0] - ls; o.y = v[q * 4 + 1] - ls;
        o.z = v[q * 4 + 2] - ls; o.w = v[q * 4 + 3] - ls;
        oi[q] = o;
    }
}

// ---------------- launcher --------------------------------------------------
extern "C" void kernel_launch(void* const* d_in, const int* in_sizes, int n_in,
                              void* d_out, int out_size, void* d_ws, size_t ws_size,
                              hipStream_t stream)
{
    const float* x  = (const float*)d_in[0];
    const float* W1 = (const float*)d_in[1];
    const float* b1 = (const float*)d_in[2];
    const float* W2 = (const float*)d_in[3];
    const float* b2 = (const float*)d_in[4];
    const int* eidx = (const int*)d_in[5];

    const int N = in_sizes[0] / F_IN;
    const int E = in_sizes[5] / 2;
    const int* row = eidx;       // destination
    const int* col = eidx + E;   // source

    char* p = (char*)d_ws;
    auto carve = [&](size_t bytes) -> void* {
        void* r = (void*)p;
        p += (bytes + 255) & ~(size_t)255;
        return r;
    };
    int*      cnt    = (int*)     carve((size_t)N * 4);
    int*      fill   = (int*)     carve((size_t)N * 4);
    int*      rowptr = (int*)     carve((size_t)(N + 1) * 4);
    float*    dinv   = (float*)   carve((size_t)N * 4);
    int*      ecol   = (int*)     carve((size_t)E * 4);
    float*    ew     = (float*)   carve((size_t)E * 4);
    float*    h      = (float*)   carve((size_t)N * CLS * 4);
    float*    zA     = (float*)   carve((size_t)N * CLS * 4);
    float*    zB     = (float*)   carve((size_t)N * CLS * 4);
    _Float16* W1T    = (_Float16*)carve((size_t)F_IN * HID * 2);
    _Float16* W2T    = (_Float16*)carve((size_t)CPAD * HID * 2);
    _Float16* h1     = (_Float16*)carve((size_t)(N + 64) * HID * 2);  // 64-row slack

    const int gN  = (N + 255) / 256;
    const int gE  = (E + 255) / 256;
    const int gNC = (N * CLS + 255) / 256;

    // weight conversion (tiny, L2-resident afterwards)
    convW1_kernel<<<(F_IN * HID + 255) / 256, 256, 0, stream>>>(W1, W1T);
    convW2_kernel<<<(CPAD * HID + 255) / 256, 256, 0, stream>>>(W2, W2T);

    // graph preprocessing -> CSR
    zero_i32<<<gN, 256, 0, stream>>>(cnt, N);
    zero_i32<<<gN, 256, 0, stream>>>(fill, N);
    count_kernel<<<gE, 256, 0, stream>>>(row, cnt, E);
    dinv_kernel<<<gN, 256, 0, stream>>>(cnt, dinv, N);
    scan_kernel<<<1, 1024, 0, stream>>>(cnt, rowptr, N);
    fill_kernel<<<gE, 256, 0, stream>>>(row, col, rowptr, fill, dinv, ecol, ew, E);

    // MLP via WMMA
    gemm1_kernel<<<(N + 31) / 32, 128, 0, stream>>>(x, W1T, b1, h1, N);
    gemm2_kernel<<<(N + 63) / 64, 128, 0, stream>>>(h1, W2T, b2, h, N);

    // APPNP propagation
    copy_f32<<<gNC, 256, 0, stream>>>(h, zA, N * CLS);
    float* zin = zA; float* zout = zB;
    for (int k = 0; k < KSTEPS; ++k) {
        step_kernel<<<gN, 256, 0, stream>>>(rowptr, ecol, ew, dinv, h, zin, zout, N);
        float* t = zin; zin = zout; zout = t;
    }
    logsoftmax_kernel<<<gN, 256, 0, stream>>>(zin, (float*)d_out, N);
}
